// BNMorph_40450001994311
// MI455X (gfx1250) — matrix operand order repaired and checked by reading, not compiled, not run on hardware
//
#include <hip/hip_runtime.h>

// ---------------- problem constants (from reference) ----------------
#define B_    8
#define H_    320
#define W_    1024
#define R_    20            // SENSE_R
#define PH    (H_ + 2*R_)   // 360  padded rows  (need r in [-20, H+19])
#define PW    (W_ + 64)     // 1088 padded cols  (need c in [-20, W+42])
#define PLANE ((size_t)H_ * W_)
#define CHSTRIDE ((size_t)PH * PW)
#define SEARCH_N 105        // (2*7+1)*(2*3+1)
#define KROWS 72            // weight rows jj = j+15, j in [-15,55], padded

typedef __attribute__((ext_vector_type(16))) _Float16 v16h;
typedef __attribute__((ext_vector_type(8)))  float    v8f;

// =====================================================================
// Kernel 1: sparsify (5x5 raster precedence) + windowed nearest search.
// Writes f16 field planes {m, m*vx, m*vy} into zero-padded workspace and
// the four non-conv outputs (orgpts/correspts) directly.
// =====================================================================
__global__ __launch_bounds__(256) void bnmorph_stage12(
    const float* __restrict__ srcMap, const float* __restrict__ dstMap,
    const float* __restrict__ xx, const float* __restrict__ yy,
    _Float16* __restrict__ ws, float* __restrict__ out)
{
  __shared__ int odx[SEARCH_N];
  __shared__ int ody[SEARCH_N];
  if (threadIdx.x < SEARCH_N) {
    odx[threadIdx.x] = (int)xx[threadIdx.x];   // sorted-by-distance offsets
    ody[threadIdx.x] = (int)yy[threadIdx.x];
  }
  __syncthreads();

  int gid = blockIdx.x * 256 + threadIdx.x;    // exact cover of B*H*W
  int x = gid % W_;
  int y = (gid / W_) % H_;
  int b = gid / (W_ * H_);
  const float* sp = srcMap + (size_t)b * PLANE;
  const float* dp = dstMap + (size_t)b * PLANE;

  bool src  = sp[(size_t)y * W_ + x] > 0.5f;
  bool kept = false;
  if (src) {
    bool pre = false;
    #pragma unroll
    for (int dy = -2; dy <= 2; ++dy)
      #pragma unroll
      for (int dx = -2; dx <= 2; ++dx) {
        if (dy < 0 || (dy == 0 && dx < 0)) {   // raster-precedes mask
          int yn = y + dy, xn = x + dx;
          if (yn >= 0 && yn < H_ && xn >= 0 && xn < W_)
            pre = pre || (sp[(size_t)yn * W_ + xn] > 0.5f);
        }
      }
    kept = !pre;
  }

  float vx = 0.f, vy = 0.f;
  bool found = false;
  if (kept) {
    for (int k = 0; k < SEARCH_N; ++k) {       // first hit == nearest
      int dx = odx[k], dy = ody[k];
      int yn = y + dy, xn = x + dx;
      if (yn >= 0 && yn < H_ && xn >= 0 && xn < W_ &&
          dp[(size_t)yn * W_ + xn] > 0.5f) {
        vx = (float)dx; vy = (float)dy; found = true; break;
      }
    }
  }
  float m = found ? 1.f : 0.f;

  // f16 field planes in padded workspace (all values exact in f16)
  size_t wbase = (size_t)(b * 3) * CHSTRIDE + (size_t)(y + R_) * PW + (x + R_);
  ws[wbase]                = (_Float16)m;
  ws[wbase + CHSTRIDE]     = (_Float16)(m * vx);
  ws[wbase + 2 * CHSTRIDE] = (_Float16)(m * vy);

  // direct (non-conv) outputs: planes 2..5 of d_out
  size_t pidx = (size_t)b * PLANE + (size_t)y * W_ + x;
  float xf = (float)x, yf = (float)y;
  out[2 * (size_t)B_ * PLANE + pidx] = xf * m;          // orgpts_x
  out[3 * (size_t)B_ * PLANE + pidx] = yf * m;          // orgpts_y
  out[4 * (size_t)B_ * PLANE + pidx] = (xf + vx) * m;   // correspts_x
  out[5 * (size_t)B_ * PLANE + pidx] = (yf + vy) * m;   // correspts_y
}

// =====================================================================
// Kernel 2: 41x41 depthwise conv of {m, m*vx, m*vy} as chained WMMAs.
// One wave per 16x16 output tile; 4 waves (128 threads) per block.
//   A[m][k] = K[jbase+m][i0+k]  (weights from row-padded LDS table,
//             rows jj = j+15 cover j in [-15,55] with zeros outside
//             [0,40], so the fragment load is branch-free)
//   B[k][n] = inp[r][x0+n+i0+k-R] (padded f16 workspace, no bounds checks)
//   C accumulates across all 56 input rows x 2 K-chunks; the output-row
//   mapping y = y0 + 15 - m is r-invariant, so one accumulator suffices.
// =====================================================================
__global__ __launch_bounds__(128) void bnmorph_conv(
    const _Float16* __restrict__ ws, float* __restrict__ out)
{
  __shared__ __align__(16) _Float16 Ksh[KROWS * 64];  // 72 rows x 64 cols

  // Analytic weight table: wts = 0.7 * exp(-sqrt(dx^2+dy^2) * 1.9/24),
  // zero outside the real 41x41 support.
  for (int i = threadIdx.x; i < KROWS * 64; i += 128) {
    int col = i & 63;
    int j   = (i >> 6) - 15;                 // true kernel row
    float v = 0.f;
    if (j >= 0 && j <= 40 && col <= 40) {
      float dx = (float)(col - R_), dy = (float)(j - R_);
      v = 0.7f * __expf(-sqrtf(dx * dx + dy * dy) * (1.9f / 24.0f));
    }
    Ksh[i] = (_Float16)v;
  }
  __syncthreads();

  int wave = threadIdx.x >> 5;
  int lane = threadIdx.x & 31;
  int tile = blockIdx.x * 4 + wave;          // 8*20*64 = 10240 tiles total
  int tx = tile % (W_ / 16);
  int ty = (tile / (W_ / 16)) % (H_ / 16);
  int b  = tile / ((W_ / 16) * (H_ / 16));
  int x0 = tx * 16, y0 = ty * 16;

  int nn      = lane & 15;                   // A row index m / B column n
  int half_hi = lane >> 4;                   // lane group 0 / 1
  int aoff    = half_hi * 8;                 // A: K 0..7,16..23 | 8..15,24..31
  int kbase   = half_hi * 8;                 // B: K 0..7,16..23 | 8..15,24..31

  v8f acc0 = {}, acc1 = {}, acc2 = {};
  const _Float16* p0 = ws + (size_t)(b * 3) * CHSTRIDE;

  #pragma unroll 1
  for (int rr = 0; rr < 56; ++rr) {          // input row r = y0 - R + rr
    // padded LDS row jj = (rr - 15 + nn) + 15 = rr + nn  -> branch-free
    const char* kb0 = (const char*)&Ksh[(rr + nn) * 64 + aoff];
    size_t rowoff = (size_t)(y0 + rr) * PW;  // padded row index (>= 0)
    __builtin_prefetch(p0 + rowoff + PW + x0, 0, 1);  // next row -> global_prefetch

    #pragma unroll
    for (int c = 0; c < 2; ++c) {
      int i0 = c * 32;
      // ---- A: weights from LDS (two aligned 16B chunks, unconditional) ----
      union { v16h h; uint4 u[2]; } A;
      A.u[0] = *reinterpret_cast<const uint4*>(kb0 + i0 * 2);
      A.u[1] = *reinterpret_cast<const uint4*>(kb0 + i0 * 2 + 32);  // +16 halfs
      // ---- B: two runs of 8 halfs (offsets 0..7, 16..23) per lane, one per channel ----
      const _Float16* r0 = p0 + rowoff + (size_t)(x0 + nn + i0 + kbase);
      v16h B0, B1, B2;
      #pragma unroll
      for (int t = 0; t < 16; ++t) {
        int tk = t < 8 ? t : t + 8;
        B0[t] = r0[tk];
        B1[t] = r0[CHSTRIDE + tk];
        B2[t] = r0[2 * CHSTRIDE + tk];
      }
      acc0 = __builtin_amdgcn_wmma_f32_16x16x32_f16(false, A.h, false, B0,
                                                    (short)0, acc0, false, false);
      acc1 = __builtin_amdgcn_wmma_f32_16x16x32_f16(false, A.h, false, B1,
                                                    (short)0, acc1, false, false);
      acc2 = __builtin_amdgcn_wmma_f32_16x16x32_f16(false, A.h, false, B2,
                                                    (short)0, acc2, false, false);
    }
  }

  // C/D layout: VGPR v -> rows m=v (lanes 0-15) and m=v+8 (lanes 16-31);
  // output row y = y0 + 15 - m, column x = x0 + (lane & 15).
  float* mx = out;
  float* my = out + (size_t)B_ * PLANE;
  #pragma unroll
  for (int v = 0; v < 8; ++v) {
    int mrow = v + half_hi * 8;
    int yo = y0 + 15 - mrow;
    int xo = x0 + nn;
    float den   = acc0[v] + 1.6f;            // ALPHA_PAD
    float dispx = acc1[v] / den;
    float dispy = acc2[v] / den;
    size_t pidx = (size_t)b * PLANE + (size_t)yo * W_ + xo;
    mx[pidx] = (float)xo + dispx;            // morphedx
    my[pidx] = (float)yo + dispy;            // morphedy
  }
}

// =====================================================================
extern "C" void kernel_launch(void* const* d_in, const int* in_sizes, int n_in,
                              void* d_out, int out_size, void* d_ws, size_t ws_size,
                              hipStream_t stream) {
  (void)in_sizes; (void)n_in; (void)out_size; (void)ws_size;
  const float* srcMap = (const float*)d_in[0];   // binMapsrc [8,1,320,1024]
  const float* dstMap = (const float*)d_in[1];   // binMapdst
  const float* xx     = (const float*)d_in[2];   // 105 sorted search offsets
  const float* yy     = (const float*)d_in[3];
  // d_in[4..7] (sxx,syy,cxx,cyy) reproduced analytically in-kernel.
  _Float16* ws  = (_Float16*)d_ws;
  float*    out = (float*)d_out;

  // zero padded f16 workspace: 3 planes * 8 batches * 360*1088 halfs (~18.8 MB)
  size_t wsBytes = (size_t)3 * B_ * CHSTRIDE * sizeof(_Float16);
  hipMemsetAsync(d_ws, 0, wsBytes, stream);

  bnmorph_stage12<<<(B_ * H_ * W_) / 256, 256, 0, stream>>>(
      srcMap, dstMap, xx, yy, ws, out);

  bnmorph_conv<<<(B_ * (H_ / 16) * (W_ / 16)) / 4, 128, 0, stream>>>(ws, out);
}
